// ScaledDotProductAttention_34437047779814
// MI455X (gfx1250) — matrix operand order repaired and checked by reading, not compiled, not run on hardware
//
#include <hip/hip_runtime.h>
#include <math.h>

// Problem constants (from reference): BH=16, S=2048, D=128
#define BHN 16
#define SQ  2048
#define DH  128
#define NKV (BHN * SQ * DH)   // elements in K (and in V)
#define SSTR 2052             // padded LDS score row stride (floats)

typedef __attribute__((ext_vector_type(16))) __bf16          v16bf;
typedef __attribute__((ext_vector_type(8)))  float           v8f;
typedef __attribute__((ext_vector_type(4)))  float           v4f;
typedef __attribute__((ext_vector_type(16))) unsigned short  v16us;
typedef __attribute__((ext_vector_type(8)))  unsigned short  v8us;
typedef __attribute__((ext_vector_type(4)))  unsigned short  v4us;

static __device__ __forceinline__ unsigned short f2bf(float x) {
    return __builtin_bit_cast(unsigned short, (__bf16)x);   // hw v_cvt
}

// ---- pre-pass A: convert f32 tensor to bf16, same layout (for K) ----
__global__ __launch_bounds__(256)
void cvt_bf16_kernel(const float* __restrict__ s, unsigned short* __restrict__ d, int n8)
{
    const int i = blockIdx.x * 256 + threadIdx.x;
    if (i >= n8) return;
    const v4f* sp = (const v4f*)s + (size_t)i * 2;
    const v4f a = sp[0], b = sp[1];
    v8us o;
    #pragma unroll
    for (int e = 0; e < 4; ++e) {
        o[e]     = f2bf(a[e]);
        o[e + 4] = f2bf(b[e]);
    }
    *((v8us*)d + i) = o;
}

// ---- pre-pass B: convert V f32 [BH,S,D] -> bf16 transposed [BH,D,S] ----
__global__ __launch_bounds__(256)
void cvt_tr_bf16_kernel(const float* __restrict__ v, unsigned short* __restrict__ vt)
{
    __shared__ unsigned short tile[32][33];
    const int b  = blockIdx.z;
    const int j0 = blockIdx.x * 32;           // seq tile
    const int d0 = blockIdx.y * 32;           // head-dim tile
    const int tx = threadIdx.x & 31;
    const int ty = threadIdx.x >> 5;          // 8 rows per pass
    const float* src = v + (size_t)b * SQ * DH;
    #pragma unroll
    for (int r = ty; r < 32; r += 8)
        tile[r][tx] = f2bf(src[(size_t)(j0 + r) * DH + d0 + tx]);
    __syncthreads();
    unsigned short* dst = vt + (size_t)b * DH * SQ;
    #pragma unroll
    for (int r = ty; r < 32; r += 8)
        dst[(size_t)(d0 + r) * SQ + j0 + tx] = tile[tx][r];
}

__global__ __launch_bounds__(128)
void sdpa_causal_kernel(const float* __restrict__ q,
                        const unsigned short* __restrict__ kbf,
                        const unsigned short* __restrict__ vt,   // [BH, D, S] bf16
                        float* __restrict__ out,
                        float* __restrict__ attn)
{
    __shared__ float          sc[16][SSTR];      // raw scores -> probabilities (~128 KB)
    __shared__ unsigned short qbf[16][DH];       // Q tile in bf16 (4 KB)
    __shared__ float          red[16][8];
    __shared__ float          rowmax[16];
    __shared__ float          rowinv[16];
    __shared__ float          ored[16][DH];      // cross-wave O reduce (8 KB)

    const int b    = blockIdx.y;
    const int it   = blockIdx.x;          // 16-row query tile index
    const int i0   = it * 16;
    const int tid  = threadIdx.x;
    const int lane = tid & 31;
    const int wave = tid >> 5;
    const int half = lane >> 4;           // lane-half
    const int l15  = lane & 15;
    const int khalfA = half * 8;          // A-matrix K sub-range base (ISA A table)
    const int khalfB = half * 16;         // contiguous-order K base; unused (B uses khalfA)

    const int ntiles = it + 1;            // causal: key tiles 0..it live
    const int jmax   = ntiles * 16;

    const float*          qrow0 = q   + ((size_t)b * SQ + i0) * DH;
    const unsigned short* kbh   = kbf + (size_t)b * SQ * DH;
    const unsigned short* vtb   = vt  + (size_t)b * DH * SQ;

    // ---- stage Q (bf16) into LDS + zero O-reduce buffer (vectorized) ----
    for (int idx = tid; idx < 16 * DH / 4; idx += 128) {
        const int m = idx >> 5, g = (idx & 31) * 4;
        const v4f qv = *(const v4f*)(qrow0 + (size_t)m * DH + g);
        v4us qb;
        #pragma unroll
        for (int e = 0; e < 4; ++e) qb[e] = f2bf(qv[e]);
        *(v4us*)&qbf[m][g]  = qb;
        *(v4f*)&ored[m][g]  = (v4f){0.f, 0.f, 0.f, 0.f};
    }
    __syncthreads();

    // contiguous key-tile range for this wave
    const int tpw = ntiles >> 2, rem = ntiles & 3;
    const int cw  = tpw + (wave < rem ? 1 : 0);
    const int sw  = wave * tpw + (wave < rem ? wave : rem);

    // =================== Q * K^T (WMMA bf16) ===================
    for (int t = sw; t < sw + cw; ++t) {
        const int j0 = t * 16;
        const int jj = j0 + l15;                          // lane's key row / col
        const unsigned short* krow = kbh + (size_t)jj * DH;
        v8f acc = {};
        #pragma unroll
        for (int c = 0; c < 4; ++c) {                     // K-dim chunks of 32 over D=128
            // A: Q rows from LDS (lane = M; elems 0-7 -> k+0..7, 8-15 -> k+16..23)
            const int offA = c * 32 + khalfA;
            v8us a0 = *(const v8us*)&qbf[l15][offA];
            v8us a1 = *(const v8us*)&qbf[l15][offA + 16];
            v16us au;
            #pragma unroll
            for (int e = 0; e < 8; ++e) { au[e] = a0[e]; au[e + 8] = a1[e]; }
            // B: K^T (lane = N key row; elems 0-7 -> k+0..7, 8-15 -> k+16..23)
            const int offB = c * 32 + khalfA;
            v8us b0 = *(const v8us*)(krow + offB);
            v8us b1 = *(const v8us*)(krow + offB + 16);
            v16us bu;
            #pragma unroll
            for (int e = 0; e < 8; ++e) { bu[e] = b0[e]; bu[e + 8] = b1[e]; }
            acc = __builtin_amdgcn_wmma_f32_16x16x32_bf16(
                      false, __builtin_bit_cast(v16bf, au),
                      false, __builtin_bit_cast(v16bf, bu),
                      (short)0, acc, false, false);
        }
        // D layout: VGPR r -> row (r + half*8), lane -> col. Mask + park in LDS.
        #pragma unroll
        for (int r = 0; r < 8; ++r) {
            const int m = r + half * 8;
            float s = acc[r];
            if (jj > i0 + m) s = -INFINITY;               // causal mask (analytic)
            sc[m][jj] = s;
        }
    }
    __syncthreads();

    // =================== softmax statistics (vectorized scans) ===================
    {
        const int m = tid >> 3, s8 = tid & 7;
        float mx = -INFINITY;
        for (int j = s8 * 4; j < jmax; j += 32) {
            const v4f x = *(const v4f*)&sc[m][j];
            mx = fmaxf(mx, fmaxf(fmaxf(x[0], x[1]), fmaxf(x[2], x[3])));
        }
        red[m][s8] = mx;
    }
    __syncthreads();
    if (tid < 16) {
        float mx = red[tid][0];
        #pragma unroll
        for (int e = 1; e < 8; ++e) mx = fmaxf(mx, red[tid][e]);
        rowmax[tid] = mx;
    }
    __syncthreads();
    {
        const int m = tid >> 3, s8 = tid & 7;
        const float mx = rowmax[m];
        float sm = 0.0f;
        for (int j = s8 * 4; j < jmax; j += 32) {
            const v4f x = *(const v4f*)&sc[m][j];
            sm += __expf(x[0] - mx) + __expf(x[1] - mx)
                + __expf(x[2] - mx) + __expf(x[3] - mx);
        }
        red[m][s8] = sm;
    }
    __syncthreads();
    if (tid < 16) {
        float sm = 0.0f;
        #pragma unroll
        for (int e = 0; e < 8; ++e) sm += red[tid][e];
        rowinv[tid] = 1.0f / sm;
    }
    __syncthreads();

    // ===== fused: normalize sc in place (P matrix) + stream P to HBM (b128) =====
    const size_t arow0 = (size_t)b * SQ * SQ + (size_t)i0 * SQ;
    for (int m = 0; m < 16; ++m) {
        const float mx = rowmax[m], inv = rowinv[m];
        for (int c = tid * 4; c < SQ; c += 128 * 4) {
            v4f val = (v4f){0.f, 0.f, 0.f, 0.f};
            if (c < jmax) {
                const v4f x = *(const v4f*)&sc[m][c];
                #pragma unroll
                for (int e = 0; e < 4; ++e) val[e] = __expf(x[e] - mx) * inv;
                *(v4f*)&sc[m][c] = val;                   // write back P
            }
            *(v4f*)(attn + arow0 + (size_t)m * SQ + c) = val;
        }
    }
    __syncthreads();

    // =================== P * V (WMMA bf16), f32 accumulate ===================
    v8f o[8];
    #pragma unroll
    for (int d = 0; d < 8; ++d) o[d] = (v8f){};

    for (int ct = sw; ct < sw + cw; ct += 2) {
        const int cbase = ct * 16;                        // 32-wide K chunk start
        const int full  = (ct + 1 < sw + cw);             // else 16-wide tail chunk
        // A: probabilities (lane = M query row) from LDS; groups of 8 contiguous
        v16us au;
        {
            const v4f p0 = *(const v4f*)&sc[l15][cbase + khalfA];
            const v4f p1 = *(const v4f*)&sc[l15][cbase + khalfA + 4];
            #pragma unroll
            for (int e = 0; e < 4; ++e) { au[e] = f2bf(p0[e]); au[e + 4] = f2bf(p1[e]); }
        }
        if (full) {                                       // wave-uniform branch
            const v4f p2 = *(const v4f*)&sc[l15][cbase + khalfA + 16];
            const v4f p3 = *(const v4f*)&sc[l15][cbase + khalfA + 20];
            #pragma unroll
            for (int e = 0; e < 4; ++e) { au[e + 8] = f2bf(p2[e]); au[e + 12] = f2bf(p3[e]); }
        } else {
            #pragma unroll
            for (int e = 8; e < 16; ++e) au[e] = 0;
        }
        const v16bf abf = __builtin_bit_cast(v16bf, au);

        #pragma unroll
        for (int dt = 0; dt < 8; ++dt) {                  // 8 tiles cover D=128
            const unsigned short* vtrow = vtb + (size_t)(dt * 16 + l15) * SQ;
            v16us bu;
            if (full) {                                   // elems 0-7 -> k+8h.., 8-15 -> k+16+8h..
                const v8us b0 = *(const v8us*)(vtrow + cbase + khalfA);
                const v8us b1 = *(const v8us*)(vtrow + cbase + khalfA + 16);
                #pragma unroll
                for (int e = 0; e < 8; ++e) { bu[e] = b0[e]; bu[e + 8] = b1[e]; }
            } else {                                      // tail: only K 0..15 live (elems 0-7)
                const v8us b0 = *(const v8us*)(vtrow + cbase + khalfA);
                #pragma unroll
                for (int e = 0; e < 8; ++e) {
                    bu[e]     = b0[e];
                    bu[e + 8] = (unsigned short)0;
                }
            }
            o[dt] = __builtin_amdgcn_wmma_f32_16x16x32_bf16(
                        false, abf,
                        false, __builtin_bit_cast(v16bf, bu),
                        (short)0, o[dt], false, false);
        }
    }

    // cross-wave reduction of O into LDS
    #pragma unroll
    for (int dt = 0; dt < 8; ++dt) {
        const int dcol = dt * 16 + l15;
        #pragma unroll
        for (int r = 0; r < 8; ++r) {
            const int m = r + half * 8;
            atomicAdd(&ored[m][dcol], o[dt][r]);
        }
    }
    __syncthreads();

    // =================== out = (P·V) * q (vectorized) ===================
    {
        const int m  = tid >> 3;                          // 16 rows, 8 threads/row
        const int c0 = (tid & 7) * 16;
        #pragma unroll
        for (int g = 0; g < 4; ++g) {
            const int d  = c0 + g * 4;
            const v4f ov = *(const v4f*)&ored[m][d];
            const v4f qv = *(const v4f*)(qrow0 + (size_t)m * DH + d);
            v4f r;
            #pragma unroll
            for (int e = 0; e < 4; ++e) r[e] = ov[e] * qv[e];
            *(v4f*)(out + ((size_t)b * SQ + i0 + m) * DH + d) = r;
        }
    }
}

extern "C" void kernel_launch(void* const* d_in, const int* in_sizes, int n_in,
                              void* d_out, int out_size, void* d_ws, size_t ws_size,
                              hipStream_t stream) {
    const float* q = (const float*)d_in[0];
    const float* k = (const float*)d_in[1];
    const float* v = (const float*)d_in[2];
    // d_in[3] is the causal bool mask; causality is applied analytically in-kernel.
    float* out  = (float*)d_out;                          // [BH, S, D]
    float* attn = out + (size_t)BHN * SQ * DH;            // [BH, S, S]

    unsigned short* kbf = (unsigned short*)d_ws;          // [BH, S, D] bf16
    unsigned short* vtb = kbf + (size_t)NKV;              // [BH, D, S] bf16 (transposed)

    const int n8 = NKV / 8;
    cvt_bf16_kernel<<<(n8 + 255) / 256, 256, 0, stream>>>(k, kbf, n8);
    dim3 tgrid(SQ / 32, DH / 32, BHN);
    cvt_tr_bf16_kernel<<<tgrid, 256, 0, stream>>>(v, vtb);

    dim3 grid(SQ / 16, BHN);
    sdpa_causal_kernel<<<grid, 128, 0, stream>>>(q, kbf, vtb, out, attn);
}
